// ModularUnit_18004502905277
// MI455X (gfx1250) — matrix operand order repaired and checked by reading, not compiled, not run on hardware
//
#include <hip/hip_runtime.h>

// ---------------------------------------------------------------------------
// LIF recurrent SNN forward scan for MI455X (gfx1250, wave32).
//   Phase 0: convert w_in, w_rec, spikes_in to f16 in workspace.
//   Phase 1: Iin = spikes @ w_in^T  via v_wmma_f32_16x16x32_f16 (time-parallel).
//   Phase 2: persistent sequential scan. Recurrent GEMV is done on the matrix
//            unit: z chunk broadcast into all 16 B columns -> D columns all
//            hold the row partials (replicated across lanes). f16 weights halve
//            the L2-resident w_rec traffic (2048 x 32MB).
// ---------------------------------------------------------------------------

#define TSTEPS 2048
#define NN     4096
#define NBLK   64                   // persistent blocks (all resident)
#define ROWS_PER_BLK (NN / NBLK)    // 64 rows per block -> 4 row-tiles of 16

#define DT_TAU_M 0.05f   // DT / tau_mem
#define DT_TAU_S 0.1f    // DT / tau_syn
#define V_LEAK  (-70.0f)
#define V_TH    (-55.0f)
#define V_RESET (-70.0f)

typedef _Float16 f16;
typedef __attribute__((ext_vector_type(16))) _Float16 v16h;
typedef __attribute__((ext_vector_type(8)))  _Float16 v8h;
typedef __attribute__((ext_vector_type(8)))  float    v8f;

// ---------------------------------------------------------------------------
// fp32 -> f16 conversion (grid-stride)
// ---------------------------------------------------------------------------
__global__ void cvt_f32_to_f16(const float* __restrict__ src,
                               f16* __restrict__ dst, int n) {
  for (int i = blockIdx.x * blockDim.x + threadIdx.x; i < n;
       i += gridDim.x * blockDim.x)
    dst[i] = (f16)src[i];
}

// ---------------------------------------------------------------------------
// zero f16 z double-buffer + barrier counter (runs every launch)
// ---------------------------------------------------------------------------
__global__ void init_state(f16* __restrict__ zbuf, unsigned* __restrict__ bar) {
  int i = blockIdx.x * blockDim.x + threadIdx.x;
  if (i < 2 * NN) zbuf[i] = (f16)0.0f;
  if (i == 0) *bar = 0u;
}

// ---------------------------------------------------------------------------
// Phase 1: Iin[t][j] = sum_k S[t,k] * Win[j,k]   (C = S @ Win^T)
// One 16x16 C tile per wave, K steps of 32 via v_wmma_f32_16x16x32_f16.
// Fragment layouts per CDNA5 ISA 7.12.2 (wave32).
// ---------------------------------------------------------------------------
__global__ void input_gemm_wmma(const f16* __restrict__ Sh,   // [T, N] f16
                                const f16* __restrict__ Wh,   // [N, N] f16
                                float* __restrict__ Iin) {    // [T, N] f32
  const int lane  = threadIdx.x & 31;
  const int wave  = threadIdx.x >> 5;
  const int tbase = blockIdx.y * 16;
  const int jbase = (blockIdx.x * 8 + wave) * 16;

  const int arow = tbase + (lane & 15);
  const int akof = (lane < 16) ? 0 : 8;
  const int bcol = jbase + (lane & 15);
  const int bkof = (lane < 16) ? 0 : 8;

  const f16* Sp = Sh + (size_t)arow * NN;
  const f16* Wp = Wh + (size_t)bcol * NN;

  v8f c = {};
  for (int kk = 0; kk < NN; kk += 32) {
    v8h alo = *(const v8h*)(Sp + kk + akof);
    v8h ahi = *(const v8h*)(Sp + kk + akof + 16);
    v16h a = __builtin_shufflevector(alo, ahi, 0, 1, 2, 3, 4, 5, 6, 7,
                                     8, 9, 10, 11, 12, 13, 14, 15);
    v8h blo = *(const v8h*)(Wp + kk + bkof);
    v8h bhi = *(const v8h*)(Wp + kk + bkof + 16);
    v16h b = __builtin_shufflevector(blo, bhi, 0, 1, 2, 3, 4, 5, 6, 7,
                                     8, 9, 10, 11, 12, 13, 14, 15);
    c = __builtin_amdgcn_wmma_f32_16x16x32_f16(
        false, a, false, b, (short)0, c, false, false);
  }

  const int mofs = (lane < 16) ? 0 : 8;
  const int col  = jbase + (lane & 15);
#pragma unroll
  for (int r = 0; r < 8; ++r)
    Iin[(size_t)(tbase + r + mofs) * NN + col] = c[r];
}

// ---------------------------------------------------------------------------
// Phase 2: persistent sequential LIF scan with WMMA recurrent GEMV.
// Block owns 64 rows = 4 row-tiles of 16. Wave w: row-tile (w&3), K-half (w>>2).
// B fragment = z chunk broadcast into all 16 columns (LDS broadcast reads):
//   D[m,n] identical over n -> lane 0 holds m=0..7, lane 16 holds m=8..15.
// ---------------------------------------------------------------------------
__global__ void lif_scan(const f16* __restrict__ w_rec_h,  // [N, N] f16
                         const float* __restrict__ Iin,    // [T, N] f32
                         f16* __restrict__ zbuf,           // [2, N] f16 dbl buffer
                         float* __restrict__ out,          // [T, N] f32 spikes
                         unsigned* __restrict__ bar) {
  __shared__ __align__(32) f16 z_lds[NN];       // 8 KB
  __shared__ float v_s[ROWS_PER_BLK];
  __shared__ float i_s[ROWS_PER_BLK];
  __shared__ float r_s[2 * ROWS_PER_BLK];       // per-K-half partials

  const int tid     = threadIdx.x;
  const int lane    = tid & 31;
  const int wave    = tid >> 5;                 // 8 waves
  const int rt      = wave & 3;                 // row-tile 0..3
  const int kh      = wave >> 2;                // K-half 0..1
  const int rowbase = blockIdx.x * ROWS_PER_BLK;

  const int arow = rowbase + rt * 16 + (lane & 15);
  const int akof = (lane < 16) ? 0 : 8;         // A-fragment K sub-offset
  const int bkof = (lane < 16) ? 0 : 8;         // B-fragment K sub-offset
  const f16* wrp = w_rec_h + (size_t)arow * NN;
  const int kk0  = kh * (NN / 2);

  if (tid < ROWS_PER_BLK) {
    v_s[tid] = V_LEAK;
    i_s[tid] = 0.0f;
  }
  __syncthreads();

  for (int t = 0; t < TSTEPS; ++t) {
    const f16* zread  = zbuf + (size_t)(t & 1) * NN;
    f16*       zwrite = zbuf + (size_t)((t & 1) ^ 1) * NN;

    // stage z_{t-1} (f16) into LDS, coalesced v8h copies
    for (int j = tid * 8; j < NN; j += 256 * 8)
      *(v8h*)&z_lds[j] = *(const v8h*)&zread[j];
    __syncthreads();

    // recurrent GEMV on the matrix unit: 64 WMMAs per wave per step
    v8f c = {};
#pragma unroll 4
    for (int kk = kk0; kk < kk0 + NN / 2; kk += 32) {
      v8h alo = *(const v8h*)(wrp + kk + akof);
      v8h ahi = *(const v8h*)(wrp + kk + akof + 16);
      v16h a = __builtin_shufflevector(alo, ahi, 0, 1, 2, 3, 4, 5, 6, 7,
                                       8, 9, 10, 11, 12, 13, 14, 15);
      // broadcast z chunk into every column of B (LDS broadcast read)
      v8h blo = *(const v8h*)&z_lds[kk + bkof];
      v8h bhi = *(const v8h*)&z_lds[kk + bkof + 16];
      v16h b = __builtin_shufflevector(blo, bhi, 0, 1, 2, 3, 4, 5, 6, 7,
                                       8, 9, 10, 11, 12, 13, 14, 15);
      c = __builtin_amdgcn_wmma_f32_16x16x32_f16(
          false, a, false, b, (short)0, c, false, false);
    }

    // D columns identical: lane 0 -> rows m=0..7, lane 16 -> rows m=8..15
    if (lane == 0) {
#pragma unroll
      for (int r = 0; r < 8; ++r) r_s[kh * 64 + rt * 16 + r] = c[r];
    } else if (lane == 16) {
#pragma unroll
      for (int r = 0; r < 8; ++r) r_s[kh * 64 + rt * 16 + 8 + r] = c[r];
    }
    __syncthreads();

    // LIF pointwise update for this block's rows (state resident in LDS)
    if (tid < ROWS_PER_BLK) {
      const int row = rowbase + tid;
      float v = v_s[tid];
      float i = i_s[tid];
      float rrec  = r_s[tid] + r_s[64 + tid];
      float v_dec = v + DT_TAU_M * ((V_LEAK - v) + i);
      float z_new = (v_dec - V_TH > 0.0f) ? 1.0f : 0.0f;
      float v_new = (z_new != 0.0f) ? V_RESET : v_dec;
      float i_new = (1.0f - DT_TAU_S) * i + Iin[(size_t)t * NN + row] + rrec;
      v_s[tid] = v_new;
      i_s[tid] = i_new;
      out[(size_t)t * NN + row] = z_new;
      zwrite[row] = (f16)z_new;   // exact: z in {0,1}
    }

    // grid-wide barrier: monotonic arrival counter, target = (t+1)*NBLK
    __syncthreads();
    if (tid == 0) {
      __threadfence();
      atomicAdd(bar, 1u);
      const unsigned target = (unsigned)(t + 1) * (unsigned)NBLK;
      while (__hip_atomic_load(bar, __ATOMIC_ACQUIRE,
                               __HIP_MEMORY_SCOPE_AGENT) < target)
        __builtin_amdgcn_s_sleep(1);
    }
    __syncthreads();
  }
}

// ---------------------------------------------------------------------------
// Host launcher. Inputs: d_in[0]=spikes_in [T*N] f32, d_in[1]=w_in [N*N] f32,
// d_in[2]=w_rec [N*N] f32. Output: [T*N] f32 spikes.
// Workspace: w_in f16 (32MB) | w_rec f16 (32MB) | spikes f16 (16MB) |
//            Iin f32 (32MB) | z f16 dbl buffer (16KB) | barrier counter
// ---------------------------------------------------------------------------
extern "C" void kernel_launch(void* const* d_in, const int* in_sizes, int n_in,
                              void* d_out, int out_size, void* d_ws, size_t ws_size,
                              hipStream_t stream) {
  const float* spikes = (const float*)d_in[0];
  const float* w_in   = (const float*)d_in[1];
  const float* w_rec  = (const float*)d_in[2];
  float* out = (float*)d_out;

  char* ws = (char*)d_ws;
  f16*   w_in_h  = (f16*)(ws);
  f16*   w_rec_h = (f16*)(ws + (size_t)NN * NN * 2);
  f16*   s_h     = (f16*)(ws + (size_t)NN * NN * 4);
  float* Iin     = (float*)(ws + (size_t)NN * NN * 4 + (size_t)TSTEPS * NN * 2);
  f16*   zbuf    = (f16*)((char*)Iin + (size_t)TSTEPS * NN * 4);
  unsigned* bar  = (unsigned*)((char*)zbuf + (size_t)2 * NN * 2);

  // Phase 0: precision conversion
  cvt_f32_to_f16<<<4096, 256, 0, stream>>>(w_in, w_in_h, NN * NN);
  cvt_f32_to_f16<<<4096, 256, 0, stream>>>(w_rec, w_rec_h, NN * NN);
  cvt_f32_to_f16<<<4096, 256, 0, stream>>>(spikes, s_h, TSTEPS * NN);
  init_state<<<(2 * NN + 255) / 256, 256, 0, stream>>>(zbuf, bar);

  // Phase 1: time-parallel input projection (WMMA GEMM)
  dim3 gemm_grid(NN / (16 * 8), TSTEPS / 16);   // (32, 128)
  input_gemm_wmma<<<gemm_grid, 256, 0, stream>>>(s_h, w_in_h, Iin);

  // Phase 2: sequential recurrent scan (persistent, grid-synced, WMMA GEMV)
  lif_scan<<<NBLK, 256, 0, stream>>>(w_rec_h, Iin, zbuf, out, bar);
}
